// LendingGAT_2808908612302
// MI455X (gfx1250) — compile-verified, matrix operand order checked by reading, not run on hardware
//
#include <hip/hip_runtime.h>
#include <math.h>

#define N_CP  100000
#define N_SEC 200000

typedef __attribute__((ext_vector_type(16))) __bf16 v16bf;
typedef __attribute__((ext_vector_type(8)))  float  v8f;

// ---------------- helpers ----------------

__device__ __forceinline__ __bf16 f2bf(float f) {
    union { float f; unsigned u; } in; in.f = f;
    unsigned u = in.u;
    unsigned r = (u + 0x7FFFu + ((u >> 16) & 1u)) >> 16;   // round-to-nearest-even
    union { unsigned short s; __bf16 b; } out; out.s = (unsigned short)r;
    return out.b;
}

__device__ __forceinline__ float eluf(float v)  { return v > 0.f ? v : expm1f(v); }

// order-preserving float<->uint encoding for atomicMax-based segment max
__device__ __forceinline__ unsigned fkey(float f) {
    unsigned u = __float_as_uint(f);
    return (u & 0x80000000u) ? ~u : (u | 0x80000000u);
}
__device__ __forceinline__ float funkey(unsigned k) {
    unsigned u = (k & 0x80000000u) ? (k & 0x7FFFFFFFu) : ~k;
    return __uint_as_float(u);
}

// ---------------- WMMA GEMM: Y[N,OUT] = X[N,K] @ W[OUT,K]^T + b ----------------
// One wave per 16-row tile; bf16 inputs, f32 accumulate via v_wmma_f32_16x16x32_bf16.
// ACT: 0 = none, 1 = ELU, 2 = ReLU.

template<int K, int OUT, int ACT>
__global__ __launch_bounds__(256)
void gemm_wmma_kernel(const float* __restrict__ X, const float* __restrict__ W,
                      const float* __restrict__ bias, float* __restrict__ Y, int nrows)
{
    const int lane = threadIdx.x & 31;
    const int wave = (int)((blockIdx.x * blockDim.x + threadIdx.x) >> 5);
    const int rowbase = wave * 16;
    if (rowbase >= nrows) return;

    const int lm = lane & 15;        // row (A) / col (B/C/D) within tile
    const int kh = lane >> 4;        // which K-half this lane holds
    constexpr int KS = (K + 31) / 32;

    // A fragments (ISA 16-bit A 16x32 layout)
    v16bf afrag[KS];
    const float* xrow = X + (size_t)(rowbase + lm) * K;
    #pragma unroll
    for (int s = 0; s < KS; ++s) {
        #pragma unroll
        for (int e = 0; e < 16; ++e) {
            int k = s * 32 + ((e < 8) ? (kh * 8 + e) : (16 + kh * 8 + (e - 8)));
            float f = (k < K) ? xrow[k] : 0.0f;
            afrag[s][e] = f2bf(f);
        }
    }

    #pragma unroll
    for (int ct = 0; ct < OUT / 16; ++ct) {
        float bv = bias[ct * 16 + lm];
        v8f acc;
        #pragma unroll
        for (int r = 0; r < 8; ++r) acc[r] = bv;

        const float* wrow = W + (size_t)(ct * 16 + lm) * K;  // W[out_col, k]
        #pragma unroll
        for (int s = 0; s < KS; ++s) {
            v16bf bfrag;
            #pragma unroll
            for (int e = 0; e < 16; ++e) {
                int k = s * 32 + ((e < 8) ? (kh * 8 + e) : (16 + kh * 8 + (e - 8)));
                float f = (k < K) ? wrow[k] : 0.0f;
                bfrag[e] = f2bf(f);
            }
            acc = __builtin_amdgcn_wmma_f32_16x16x32_bf16(
                /*neg_a=*/false, afrag[s], /*neg_b=*/false, bfrag,
                /*c_mod=*/(short)0, acc, /*reuse_a=*/false, /*reuse_b=*/false);
        }

        // C/D layout: VGPR r holds row (r + kh*8), column lm
        #pragma unroll
        for (int r = 0; r < 8; ++r) {
            float v = acc[r];
            if (ACT == 1)      v = eluf(v);
            else if (ACT == 2) v = fmaxf(v, 0.f);
            Y[(size_t)(rowbase + kh * 8 + r) * OUT + ct * 16 + lm] = v;
        }
    }
}

// ---------------- edge kernels (GATv2 attention) ----------------
// H=4 heads, C=16 channels, HID=64. att is [4,16] flat.

template<bool SELF>
__global__ __launch_bounds__(256)
void edge_max_kernel(const int* __restrict__ esrc, const int* __restrict__ edst, int E,
                     const float* __restrict__ xl, const float* __restrict__ xr,
                     const float* __restrict__ att, unsigned* __restrict__ mkey)
{
    int e = blockIdx.x * blockDim.x + threadIdx.x;
    if (e >= E) return;
    int s = SELF ? e : esrc[e];
    int d = SELF ? e : edst[e];
    const float* pl = xl + (size_t)s * 64;
    const float* pr = xr + (size_t)d * 64;
    __builtin_prefetch(pl, 0, 1);
    __builtin_prefetch(pr, 0, 1);
    float logit[4] = {0.f, 0.f, 0.f, 0.f};
    #pragma unroll
    for (int c = 0; c < 64; ++c) {
        float v = pl[c] + pr[c];
        v = (v > 0.f) ? v : 0.2f * v;            // leaky_relu(0.2)
        logit[c >> 4] += v * att[c];
    }
    #pragma unroll
    for (int h = 0; h < 4; ++h)
        atomicMax(&mkey[(size_t)d * 4 + h], fkey(logit[h]));
}

template<bool SELF>
__global__ __launch_bounds__(256)
void edge_accum_kernel(const int* __restrict__ esrc, const int* __restrict__ edst, int E,
                       const float* __restrict__ xl, const float* __restrict__ xr,
                       const float* __restrict__ att, const unsigned* __restrict__ mkey,
                       float* __restrict__ denom, float* __restrict__ outb)
{
    int e = blockIdx.x * blockDim.x + threadIdx.x;
    if (e >= E) return;
    int s = SELF ? e : esrc[e];
    int d = SELF ? e : edst[e];
    const float* pl = xl + (size_t)s * 64;
    const float* pr = xr + (size_t)d * 64;
    __builtin_prefetch(pl, 0, 1);
    __builtin_prefetch(pr, 0, 1);
    float logit[4] = {0.f, 0.f, 0.f, 0.f};
    #pragma unroll
    for (int c = 0; c < 64; ++c) {
        float v = pl[c] + pr[c];
        v = (v > 0.f) ? v : 0.2f * v;
        logit[c >> 4] += v * att[c];
    }
    float a[4];
    #pragma unroll
    for (int h = 0; h < 4; ++h) {
        a[h] = __expf(logit[h] - funkey(mkey[(size_t)d * 4 + h]));
        atomicAdd(&denom[(size_t)d * 4 + h], a[h]);
    }
    float* po = outb + (size_t)d * 64;
    #pragma unroll
    for (int c = 0; c < 64; ++c)
        atomicAdd(&po[c], a[c >> 4] * pl[c]);
}

// ---------------- finalize / misc ----------------

__global__ __launch_bounds__(256)
void fill_kernel(unsigned* __restrict__ p, unsigned val, size_t n)
{
    size_t i = (size_t)blockIdx.x * blockDim.x + threadIdx.x;
    if (i < n) p[i] = val;
}

// h_new = elu(segsum/denom + bias)  (isolated dst -> bias only)
__global__ __launch_bounds__(256)
void finalize_one_kernel(const float* __restrict__ outb, const float* __restrict__ den,
                         const float* __restrict__ bias, float* __restrict__ hnew, int n)
{
    int i = blockIdx.x * blockDim.x + threadIdx.x;
    if (i >= n * 64) return;
    int node = i >> 6, c = i & 63;
    float dv = den[(size_t)node * 4 + (c >> 4)];
    float v = (dv > 0.f) ? outb[i] / dv : 0.f;
    hnew[i] = eluf(v + bias[c]);
}

// h_new = elu( gat1 + gat2 )   (HeteroConv aggr='sum')
__global__ __launch_bounds__(256)
void finalize_sum_kernel(const float* __restrict__ o1, const float* __restrict__ d1,
                         const float* __restrict__ b1,
                         const float* __restrict__ o2, const float* __restrict__ d2,
                         const float* __restrict__ b2,
                         float* __restrict__ hnew, int n)
{
    int i = blockIdx.x * blockDim.x + threadIdx.x;
    if (i >= n * 64) return;
    int node = i >> 6, c = i & 63;
    float dv1 = d1[(size_t)node * 4 + (c >> 4)];
    float dv2 = d2[(size_t)node * 4 + (c >> 4)];
    float v1 = (dv1 > 0.f) ? o1[i] / dv1 : 0.f;
    float v2 = (dv2 > 0.f) ? o2[i] / dv2 : 0.f;
    hnew[i] = eluf(v1 + b1[c] + v2 + b2[c]);
}

__global__ __launch_bounds__(256)
void score_kernel(const float* __restrict__ hid, const float* __restrict__ W2,
                  const float* __restrict__ b2, float* __restrict__ out, int n)
{
    int i = blockIdx.x * blockDim.x + threadIdx.x;
    if (i >= n) return;
    float s = b2[0];
    const float* row = hid + (size_t)i * 32;
    #pragma unroll
    for (int j = 0; j < 32; ++j) s += row[j] * W2[j];
    out[i] = 1.f / (1.f + __expf(-s));
}

__global__ __launch_bounds__(256)
void copy_kernel(const float* __restrict__ src, float* __restrict__ dst, size_t n)
{
    size_t i = (size_t)blockIdx.x * blockDim.x + threadIdx.x;
    if (i < n) dst[i] = src[i];
}

// ---------------- host driver ----------------

static inline int gn(size_t n) { return (int)((n + 255) / 256); }
static inline int gw(int rows) { return (int)(((size_t)(rows / 16) * 32 + 255) / 256); }

extern "C" void kernel_launch(void* const* d_in, const int* in_sizes, int n_in,
                              void* d_out, int out_size, void* d_ws, size_t ws_size,
                              hipStream_t stream)
{
    (void)n_in; (void)out_size; (void)ws_size;

    const float* x_cp  = (const float*)d_in[0];
    const float* x_sec = (const float*)d_in[1];
    const int*   eb    = (const int*)d_in[2];
    const int*   ep    = (const int*)d_in[3];
    const int*   ec    = (const int*)d_in[4];
    const int E_B = in_sizes[2] / 2;
    const int E_P = in_sizes[3] / 2;
    const int E_C = in_sizes[4] / 2;

    // params flattened in setup_inputs() insertion order:
    // 5: cp_proj_W  6: cp_proj_b  7: sec_proj_W  8: sec_proj_b
    // 9 + l*18 + {borrows:0, pledged:6, co:12} + {W_l:0, b_l:1, W_r:2, b_r:3, att:4, bias:5}
    // 63: h1_W  64: h1_b  65: h2_W  66: h2_b
    auto PF = [&](int i) { return (const float*)d_in[i]; };

    // workspace layout (floats)
    float* w = (float*)d_ws;
    size_t off = 0;
    auto alloc = [&](size_t nf) { float* p = w + off; off += nf; return p; };
    float*    hcpA    = alloc((size_t)N_CP * 64);
    float*    hcpB    = alloc((size_t)N_CP * 64);
    float*    hsecA   = alloc((size_t)N_SEC * 64);
    float*    hsecB   = alloc((size_t)N_SEC * 64);
    float*    xl      = alloc((size_t)N_SEC * 64);
    float*    xr      = alloc((size_t)N_SEC * 64);
    float*    outsec  = alloc((size_t)N_SEC * 64);
    float*    outp    = alloc((size_t)N_CP * 64);
    float*    outc    = alloc((size_t)N_CP * 64);
    unsigned* mkey    = (unsigned*)alloc((size_t)N_SEC * 4);
    float*    den_sec = alloc((size_t)N_SEC * 4);
    float*    den_p   = alloc((size_t)N_CP * 4);
    float*    den_c   = alloc((size_t)N_CP * 4);
    float*    hidb    = alloc((size_t)N_CP * 32);

    // input projections (ELU fused)
    gemm_wmma_kernel<32, 64, 1><<<gw(N_CP),  256, 0, stream>>>(x_cp,  PF(5), PF(6), hcpA,  N_CP);
    gemm_wmma_kernel<16, 64, 1><<<gw(N_SEC), 256, 0, stream>>>(x_sec, PF(7), PF(8), hsecA, N_SEC);

    float* hcp = hcpA;  float* hcp_n = hcpB;
    float* hsec = hsecA; float* hsec_n = hsecB;

    for (int l = 0; l < 3; ++l) {
        const int Pb = 9 + l * 18;      // borrows
        const int Pp = Pb + 6;          // pledged
        const int Pc = Pb + 12;         // co

        // -------- borrows: cp -> sec --------
        gemm_wmma_kernel<64, 64, 0><<<gw(N_CP),  256, 0, stream>>>(hcp,  PF(Pb + 0), PF(Pb + 1), xl, N_CP);
        gemm_wmma_kernel<64, 64, 0><<<gw(N_SEC), 256, 0, stream>>>(hsec, PF(Pb + 2), PF(Pb + 3), xr, N_SEC);
        fill_kernel<<<gn((size_t)N_SEC * 4),  256, 0, stream>>>(mkey, 0u, (size_t)N_SEC * 4);
        fill_kernel<<<gn((size_t)N_SEC * 4),  256, 0, stream>>>((unsigned*)den_sec, 0u, (size_t)N_SEC * 4);
        fill_kernel<<<gn((size_t)N_SEC * 64), 256, 0, stream>>>((unsigned*)outsec, 0u, (size_t)N_SEC * 64);
        edge_max_kernel<false><<<gn(E_B), 256, 0, stream>>>(eb, eb + E_B, E_B, xl, xr, PF(Pb + 4), mkey);
        edge_accum_kernel<false><<<gn(E_B), 256, 0, stream>>>(eb, eb + E_B, E_B, xl, xr, PF(Pb + 4), mkey, den_sec, outsec);
        finalize_one_kernel<<<gn((size_t)N_SEC * 64), 256, 0, stream>>>(outsec, den_sec, PF(Pb + 5), hsec_n, N_SEC);

        // -------- pledged: sec -> cp --------
        gemm_wmma_kernel<64, 64, 0><<<gw(N_SEC), 256, 0, stream>>>(hsec, PF(Pp + 0), PF(Pp + 1), xl, N_SEC);
        gemm_wmma_kernel<64, 64, 0><<<gw(N_CP),  256, 0, stream>>>(hcp,  PF(Pp + 2), PF(Pp + 3), xr, N_CP);
        fill_kernel<<<gn((size_t)N_CP * 4),  256, 0, stream>>>(mkey, 0u, (size_t)N_CP * 4);
        fill_kernel<<<gn((size_t)N_CP * 4),  256, 0, stream>>>((unsigned*)den_p, 0u, (size_t)N_CP * 4);
        fill_kernel<<<gn((size_t)N_CP * 64), 256, 0, stream>>>((unsigned*)outp, 0u, (size_t)N_CP * 64);
        edge_max_kernel<false><<<gn(E_P), 256, 0, stream>>>(ep, ep + E_P, E_P, xl, xr, PF(Pp + 4), mkey);
        edge_accum_kernel<false><<<gn(E_P), 256, 0, stream>>>(ep, ep + E_P, E_P, xl, xr, PF(Pp + 4), mkey, den_p, outp);

        // -------- co: cp -> cp (+ implicit self loops) --------
        gemm_wmma_kernel<64, 64, 0><<<gw(N_CP), 256, 0, stream>>>(hcp, PF(Pc + 0), PF(Pc + 1), xl, N_CP);
        gemm_wmma_kernel<64, 64, 0><<<gw(N_CP), 256, 0, stream>>>(hcp, PF(Pc + 2), PF(Pc + 3), xr, N_CP);
        fill_kernel<<<gn((size_t)N_CP * 4),  256, 0, stream>>>(mkey, 0u, (size_t)N_CP * 4);
        fill_kernel<<<gn((size_t)N_CP * 4),  256, 0, stream>>>((unsigned*)den_c, 0u, (size_t)N_CP * 4);
        fill_kernel<<<gn((size_t)N_CP * 64), 256, 0, stream>>>((unsigned*)outc, 0u, (size_t)N_CP * 64);
        edge_max_kernel<false><<<gn(E_C),  256, 0, stream>>>(ec, ec + E_C, E_C, xl, xr, PF(Pc + 4), mkey);
        edge_max_kernel<true><<<gn(N_CP),  256, 0, stream>>>(nullptr, nullptr, N_CP, xl, xr, PF(Pc + 4), mkey);
        edge_accum_kernel<false><<<gn(E_C), 256, 0, stream>>>(ec, ec + E_C, E_C, xl, xr, PF(Pc + 4), mkey, den_c, outc);
        edge_accum_kernel<true><<<gn(N_CP), 256, 0, stream>>>(nullptr, nullptr, N_CP, xl, xr, PF(Pc + 4), mkey, den_c, outc);

        finalize_sum_kernel<<<gn((size_t)N_CP * 64), 256, 0, stream>>>(
            outp, den_p, PF(Pp + 5), outc, den_c, PF(Pc + 5), hcp_n, N_CP);

        float* t;
        t = hcp;  hcp  = hcp_n;  hcp_n  = t;
        t = hsec; hsec = hsec_n; hsec_n = t;
    }

    // prediction head
    gemm_wmma_kernel<64, 32, 2><<<gw(N_CP), 256, 0, stream>>>(hcp, PF(63), PF(64), hidb, N_CP);
    score_kernel<<<gn(N_CP), 256, 0, stream>>>(hidb, PF(65), PF(66), (float*)d_out, N_CP);
    copy_kernel<<<gn((size_t)N_CP * 64), 256, 0, stream>>>(hcp, (float*)d_out + N_CP, (size_t)N_CP * 64);
}
